// RichAttention_515396075768
// MI455X (gfx1250) — compile-verified, matrix operand order checked by reading, not run on hardware
//
#include <hip/hip_runtime.h>
#include <hip/hip_bf16.h>

// RichAttention for MI455X (gfx1250, wave32, WMMA).
// d_out = [out (B*S*H f32)] ++ [attn (B*S*S f32)]
// d_ws  = q_f16, k_f16, v_f16  (3 * B*S*H halves = 75.5 MB)
//
// GEMM scheme (all three matmul kernels):
//   block tile 64(M) x 256(N), BK=32, 256 threads = 8 waves (2x4),
//   wave tile 32x64 = 2x4 WMMA tiles -> 8 v_wmma per K-step from 6 frag loads.
//   Double-buffered LDS staging: one barrier per K-step, stage(k+1) overlaps wmma(k).

typedef __attribute__((ext_vector_type(16))) _Float16 v16h;
typedef __attribute__((ext_vector_type(8)))  _Float16 v8h;
typedef __attribute__((ext_vector_type(4)))  _Float16 v4h;
typedef __attribute__((ext_vector_type(8)))  float    v8f;

#define B_  8
#define S_  2048
#define H_  768
#define BK  32
#define BM  64
#define BN  256
#define LDSTR 40   // LDS row stride in halves (32 + 8 pad -> conflict-free b128 frag loads)

// ---- fragment load from LDS tile (16 rows starting at `t`, row stride LDSTR halves) ----
// CDNA5 16-bit 16x32 A/B layout: lane L holds row/col (L&15); halves 0..7 hold
// K = kb..kb+7 and halves 8..15 hold K = kb+16..kb+23, kb = 8*(L>=16).
__device__ __forceinline__ v16h frag_ld(const _Float16* t, int lane) {
  const int r  = lane & 15;
  const int kb = (lane >> 4) << 3;
  const _Float16* p = t + r * LDSTR + kb;
  v8h lo = *(const v8h*)(p);
  v8h hi = *(const v8h*)(p + 16);
  v16h f;
#pragma unroll
  for (int i = 0; i < 8; ++i) { f[i] = lo[i]; f[i + 8] = hi[i]; }
  return f;
}

__device__ __forceinline__ v8f wmma_f16(v16h a, v16h b, v8f c) {
  return __builtin_amdgcn_wmma_f32_16x16x32_f16(false, a, false, b, (short)0, c, false, false);
}

// ============================================================================
// Kernel 1: fused QKV projection. Y = X @ W^T + bias, stored f16.
// grid = (M/64, 768/256=3, 3)
// ============================================================================
__global__ void __launch_bounds__(256)
proj_kernel(const float* __restrict__ xq, const float* __restrict__ xk,
            const float* __restrict__ xv,
            const float* __restrict__ Wq, const float* __restrict__ bq,
            const float* __restrict__ Wk, const float* __restrict__ bk,
            const float* __restrict__ Wv, const float* __restrict__ bv,
            _Float16* __restrict__ qf, _Float16* __restrict__ kf,
            _Float16* __restrict__ vf) {
  __shared__ _Float16 lds_a[2][BM][LDSTR];
  __shared__ _Float16 lds_b[2][BN][LDSTR];

  const float* X; const float* W; const float* bias; _Float16* Y;
  switch (blockIdx.z) {
    case 0:  X = xq; W = Wq; bias = bq; Y = qf; break;
    case 1:  X = xk; W = Wk; bias = bk; Y = kf; break;
    default: X = xv; W = Wv; bias = bv; Y = vf; break;
  }

  const int tid  = threadIdx.x;
  const int lane = tid & 31;
  const int wave = tid >> 5;
  const int wm = wave >> 2, wn = wave & 3;
  const int m0 = blockIdx.x * BM;
  const int n0 = blockIdx.y * BN;

  auto stage = [&](int buf, int k0) {
#pragma unroll
    for (int i = 0; i < 2; ++i) {            // A: 64x32 f32 -> f16
      int c = tid + i * 256;
      int row = c >> 3, k4 = (c & 7) << 2;
      const float* src = X + (size_t)(m0 + row) * H_ + k0 + k4;
      if (k0 + BK < H_) __builtin_prefetch((const void*)(src + BK), 0, 1);
      float4 f = *(const float4*)src;
      v4h h = {(_Float16)f.x, (_Float16)f.y, (_Float16)f.z, (_Float16)f.w};
      *(v4h*)&lds_a[buf][row][k4] = h;
    }
#pragma unroll
    for (int i = 0; i < 8; ++i) {            // B = W rows: 256x32 f32 -> f16
      int c = tid + i * 256;
      int row = c >> 3, k4 = (c & 7) << 2;
      float4 f = *(const float4*)(W + (size_t)(n0 + row) * H_ + k0 + k4);
      v4h h = {(_Float16)f.x, (_Float16)f.y, (_Float16)f.z, (_Float16)f.w};
      *(v4h*)&lds_b[buf][row][k4] = h;
    }
  };

  v8f acc[2][4] = {};
  stage(0, 0);
  const int nk = H_ / BK;
  for (int kt = 0; kt < nk; ++kt) {
    __syncthreads();
    const int cur = kt & 1;
    if (kt + 1 < nk) stage(cur ^ 1, (kt + 1) * BK);
    v16h a0 = frag_ld(&lds_a[cur][wm * 32][0], lane);
    v16h a1 = frag_ld(&lds_a[cur][wm * 32 + 16][0], lane);
#pragma unroll
    for (int j = 0; j < 4; ++j) {
      v16h bj = frag_ld(&lds_b[cur][wn * 64 + j * 16][0], lane);
      acc[0][j] = wmma_f16(a0, bj, acc[0][j]);
      acc[1][j] = wmma_f16(a1, bj, acc[1][j]);
    }
  }

  const int col = lane & 15;
  const int rhi = (lane >> 4) << 3;
#pragma unroll
  for (int tm = 0; tm < 2; ++tm)
#pragma unroll
    for (int tn = 0; tn < 4; ++tn) {
      int nn = n0 + wn * 64 + tn * 16 + col;
      float bval = bias[nn];
#pragma unroll
      for (int i = 0; i < 8; ++i) {
        int mm = m0 + wm * 32 + tm * 16 + rhi + i;
        Y[(size_t)mm * H_ + nn] = (_Float16)(acc[tm][tn][i] + bval);
      }
    }
}

// ============================================================================
// Kernel 2: scores = q @ k^T - w * dist(coords).  grid = (S/64, S/256=8, B)
// ============================================================================
__global__ void __launch_bounds__(256)
scores_kernel(const _Float16* __restrict__ qf, const _Float16* __restrict__ kf,
              const float* __restrict__ coords, const float* __restrict__ sw,
              float* __restrict__ attn) {
  __shared__ _Float16 lds_a[2][BM][LDSTR];
  __shared__ _Float16 lds_b[2][BN][LDSTR];
  __shared__ float2 s_cs[BM];
  __shared__ float2 s_ct[BN];

  const int tid  = threadIdx.x;
  const int lane = tid & 31;
  const int wave = tid >> 5;
  const int wm = wave >> 2, wn = wave & 3;
  const int b  = blockIdx.z;
  const int m0 = blockIdx.x * BM;
  const int n0 = blockIdx.y * BN;

  const _Float16* Q = qf + (size_t)b * S_ * H_;
  const _Float16* K = kf + (size_t)b * S_ * H_;
  const float2*   C = (const float2*)coords + (size_t)b * S_;

  if (tid < BM) s_cs[tid] = C[m0 + tid];
  s_ct[tid] = C[n0 + tid];

  auto stage = [&](int buf, int k0) {
    {                                        // A: 64x32 f16, one b128/thread
      int row = tid >> 2, kk = (tid & 3) << 3;
      const _Float16* src = Q + (size_t)(m0 + row) * H_ + k0 + kk;
      if (k0 + BK < H_) __builtin_prefetch((const void*)(src + BK), 0, 1);
      *(v8h*)&lds_a[buf][row][kk] = *(const v8h*)src;
    }
#pragma unroll
    for (int i = 0; i < 4; ++i) {            // B: 256x32 f16
      int c = tid + i * 256;
      int row = c >> 2, kk = (c & 3) << 3;
      *(v8h*)&lds_b[buf][row][kk] = *(const v8h*)(K + (size_t)(n0 + row) * H_ + k0 + kk);
    }
  };

  v8f acc[2][4] = {};
  stage(0, 0);
  const int nk = H_ / BK;
  for (int kt = 0; kt < nk; ++kt) {
    __syncthreads();
    const int cur = kt & 1;
    if (kt + 1 < nk) stage(cur ^ 1, (kt + 1) * BK);
    v16h a0 = frag_ld(&lds_a[cur][wm * 32][0], lane);
    v16h a1 = frag_ld(&lds_a[cur][wm * 32 + 16][0], lane);
#pragma unroll
    for (int j = 0; j < 4; ++j) {
      v16h bj = frag_ld(&lds_b[cur][wn * 64 + j * 16][0], lane);
      acc[0][j] = wmma_f16(a0, bj, acc[0][j]);
      acc[1][j] = wmma_f16(a1, bj, acc[1][j]);
    }
  }

  const float w = sw[0];
  float* out = attn + (size_t)b * S_ * S_;
  const int col = lane & 15;
  const int rhi = (lane >> 4) << 3;
#pragma unroll
  for (int tm = 0; tm < 2; ++tm)
#pragma unroll
    for (int tn = 0; tn < 4; ++tn) {
      int nl = wn * 64 + tn * 16 + col;
      float2 ct = s_ct[nl];
#pragma unroll
      for (int i = 0; i < 8; ++i) {
        int ml = wm * 32 + tm * 16 + rhi + i;
        float2 cs = s_cs[ml];
        float dx = cs.x - ct.x, dy = cs.y - ct.y;
        float d = sqrtf(fmaxf(dx * dx + dy * dy, 1e-12f));
        out[(size_t)(m0 + ml) * S_ + n0 + nl] = acc[tm][tn][i] - w * d;
      }
    }
}

// ============================================================================
// Kernel 3: row softmax in place.  grid = B*S blocks, 256 threads, 8 elems/thread
// ============================================================================
__global__ void __launch_bounds__(256)
softmax_kernel(float* __restrict__ attn) {
  __shared__ float red[256];
  const int tid = threadIdx.x;
  float* row = attn + (size_t)blockIdx.x * S_;

  float x[8];
  float m = -3.4e38f;
#pragma unroll
  for (int j = 0; j < 8; ++j) { x[j] = row[tid + j * 256]; m = fmaxf(m, x[j]); }
  red[tid] = m;
  __syncthreads();
  for (int s = 128; s > 0; s >>= 1) {
    if (tid < s) red[tid] = fmaxf(red[tid], red[tid + s]);
    __syncthreads();
  }
  m = red[0];
  __syncthreads();

  float sum = 0.f;
#pragma unroll
  for (int j = 0; j < 8; ++j) { x[j] = __expf(x[j] - m); sum += x[j]; }
  red[tid] = sum;
  __syncthreads();
  for (int s = 128; s > 0; s >>= 1) {
    if (tid < s) red[tid] += red[tid + s];
    __syncthreads();
  }
  float inv = 1.0f / red[0];
#pragma unroll
  for (int j = 0; j < 8; ++j) row[tid + j * 256] = x[j] * inv;
}

// ============================================================================
// Kernel 4: out = attn @ v.  grid = (S/64, 768/256=3, B).
// A = attn f32 (converted on stage), B = v f16 transposed into LDS [h][t].
// ============================================================================
__global__ void __launch_bounds__(256)
out_kernel(const float* __restrict__ attn, const _Float16* __restrict__ vf,
           float* __restrict__ out) {
  __shared__ _Float16 lds_a[2][BM][LDSTR];
  __shared__ _Float16 lds_b[2][BN][LDSTR];

  const int tid  = threadIdx.x;
  const int lane = tid & 31;
  const int wave = tid >> 5;
  const int wm = wave >> 2, wn = wave & 3;
  const int b  = blockIdx.z;
  const int m0 = blockIdx.x * BM;
  const int n0 = blockIdx.y * BN;

  const float*    A = attn + (size_t)b * S_ * S_;
  const _Float16* V = vf + (size_t)b * S_ * H_;

  auto stage = [&](int buf, int k0) {
#pragma unroll
    for (int i = 0; i < 2; ++i) {            // A: 64x32 f32 -> f16
      int c = tid + i * 256;
      int row = c >> 3, k4 = (c & 7) << 2;
      const float* src = A + (size_t)(m0 + row) * S_ + k0 + k4;
      if (k0 + BK < S_) __builtin_prefetch((const void*)(src + BK), 0, 1);
      float4 f = *(const float4*)src;
      v4h h = {(_Float16)f.x, (_Float16)f.y, (_Float16)f.z, (_Float16)f.w};
      *(v4h*)&lds_a[buf][row][k4] = h;
    }
    {                                        // B: v[t][h] -> lds_b[h][t], 32 halves/thread
      int kt = tid >> 3;                     // 0..31
      int hs = (tid & 7) << 5;               // 0..224
      const _Float16* src = V + (size_t)(k0 + kt) * H_ + n0 + hs;
#pragma unroll
      for (int q = 0; q < 4; ++q) {
        v8h p = *(const v8h*)(src + q * 8);
#pragma unroll
        for (int j = 0; j < 8; ++j) lds_b[buf][hs + q * 8 + j][kt] = p[j];
      }
    }
  };

  v8f acc[2][4] = {};
  stage(0, 0);
  const int nk = S_ / BK;
  for (int kt = 0; kt < nk; ++kt) {
    __syncthreads();
    const int cur = kt & 1;
    if (kt + 1 < nk) stage(cur ^ 1, (kt + 1) * BK);
    v16h a0 = frag_ld(&lds_a[cur][wm * 32][0], lane);
    v16h a1 = frag_ld(&lds_a[cur][wm * 32 + 16][0], lane);
#pragma unroll
    for (int j = 0; j < 4; ++j) {
      v16h bj = frag_ld(&lds_b[cur][wn * 64 + j * 16][0], lane);
      acc[0][j] = wmma_f16(a0, bj, acc[0][j]);
      acc[1][j] = wmma_f16(a1, bj, acc[1][j]);
    }
  }

  const int col = lane & 15;
  const int rhi = (lane >> 4) << 3;
#pragma unroll
  for (int tm = 0; tm < 2; ++tm)
#pragma unroll
    for (int tn = 0; tn < 4; ++tn) {
      int nn = n0 + wn * 64 + tn * 16 + col;
#pragma unroll
      for (int i = 0; i < 8; ++i) {
        int mm = m0 + wm * 32 + tm * 16 + rhi + i;
        out[((size_t)b * S_ + mm) * H_ + nn] = acc[tm][tn][i];
      }
    }
}

// ============================================================================
extern "C" void kernel_launch(void* const* d_in, const int* in_sizes, int n_in,
                              void* d_out, int out_size, void* d_ws, size_t ws_size,
                              hipStream_t stream) {
  (void)in_sizes; (void)n_in; (void)out_size; (void)ws_size;
  const float* queries = (const float*)d_in[0];
  const float* keys    = (const float*)d_in[1];
  const float* values  = (const float*)d_in[2];
  const float* coords  = (const float*)d_in[3];
  const float* Wq = (const float*)d_in[4];
  const float* bq = (const float*)d_in[5];
  const float* Wk = (const float*)d_in[6];
  const float* bk = (const float*)d_in[7];
  const float* Wv = (const float*)d_in[8];
  const float* bv = (const float*)d_in[9];
  const float* sw = (const float*)d_in[10];

  float* outp = (float*)d_out;                         // [B,S,H]
  float* attn = outp + (size_t)B_ * S_ * H_;           // [B,S,S]

  _Float16* qf = (_Float16*)d_ws;                      // 3*B*S*H halves = 75.5 MB
  _Float16* kf = qf + (size_t)B_ * S_ * H_;
  _Float16* vf = kf + (size_t)B_ * S_ * H_;

  proj_kernel<<<dim3((B_ * S_) / BM, H_ / BN, 3), 256, 0, stream>>>(
      queries, keys, values, Wq, bq, Wk, bk, Wv, bv, qf, kf, vf);

  scores_kernel<<<dim3(S_ / BM, S_ / BN, B_), 256, 0, stream>>>(
      qf, kf, coords, sw, attn);

  softmax_kernel<<<dim3(B_ * S_), 256, 0, stream>>>(attn);

  out_kernel<<<dim3(S_ / BM, H_ / BN, B_), 256, 0, stream>>>(attn, vf, outp);
}
